// HyenaFilter_53712861003971
// MI455X (gfx1250) — matrix operand order repaired and checked by reading, not compiled, not run on hardware
//
#include <hip/hip_runtime.h>
#include <hip/hip_bf16.h>
#include <math.h>
#include <stdint.h>

// ---------------------------------------------------------------------------
// Hyena FFT long-convolution for MI455X (gfx1250).
//   filter MLP  -> v_wmma_f32_16x16x32_bf16 (wave32 fragment layouts)
//   convolution -> 8192-pt complex FFT in LDS, 2 real channels per transform,
//                  async memory<->LDS staging (ASYNCcnt) for x and out.
// 1/N inverse-FFT scale is folded into the filter spectra so the final
// output is a pure LDS -> global async copy.
// ---------------------------------------------------------------------------

typedef __attribute__((ext_vector_type(16))) __bf16 v16bf;
typedef __attribute__((ext_vector_type(8)))  float  v8f;

#define DIM_D 1024
#define SEQ_L 4096
#define FFT_N 8192

__device__ __forceinline__ unsigned brev13(unsigned l) { return __brev(l) >> 19; }

// K offset inside a bf16 16x32 A fragment for element e (0..15), lane-half h.
__device__ __forceinline__ int koffA(int e, int h) {
    int v = e >> 1, s = e & 1;
    return (v < 4 ? 2 * v + s : 16 + 2 * (v - 4) + s) + 8 * h;
}
// K offset inside a bf16 32x16 B fragment.
__device__ __forceinline__ int koffB(int e, int h) {
    int v = e >> 1, s = e & 1;
    return (v < 4 ? 2 * v + s : 16 + 2 * (v - 4) + s) + 8 * h;
}

// ---------------------------------------------------------------------------
// Stage 1: H = GELU(PE @ W1 + b1), PE computed analytically. H stored bf16.
// One wave per 16x16 output tile; block = 4 waves covers 16(M) x 64(N).
// ---------------------------------------------------------------------------
__global__ __launch_bounds__(128)
void filter_mlp1(const float* __restrict__ W1, const float* __restrict__ b1,
                 __bf16* __restrict__ H) {
    const int lane = threadIdx.x & 31;
    const int wave = threadIdx.x >> 5;
    const int h = lane >> 4, ln = lane & 15;
    const int n0 = blockIdx.x * 64 + wave * 16;
    const int m0 = blockIdx.y * 16;
    const int l  = m0 + ln;                 // A-fragment row (M = lane & 15)

    v8f c = {};
    #pragma unroll
    for (int chunk = 0; chunk < 2; ++chunk) {
        v16bf a, b;
        #pragma unroll
        for (int e = 0; e < 16; ++e) {
            // A: sinusoidal positional embedding, dim=64
            int kA = chunk * 32 + koffA(e, h);
            int i  = kA >> 1;               // frequency index
            float ang = (float)l * __expf(-0.28782313662425574f * (float)i);
            float pe  = (kA & 1) ? __cosf(ang) : __sinf(ang);
            a[e] = (__bf16)pe;
            // B: W1 is (64 x D) row-major
            int kB = chunk * 32 + koffB(e, h);
            b[e] = (__bf16)W1[kB * DIM_D + n0 + ln];
        }
        c = __builtin_amdgcn_wmma_f32_16x16x32_bf16(false, a, false, b,
                                                    (short)0, c, false, false);
    }
    const float bias = b1[n0 + ln];
    #pragma unroll
    for (int r = 0; r < 8; ++r) {
        int m = m0 + r + 8 * h;
        float xv = c[r] + bias;
        float g  = 0.5f * xv * (1.0f + erff(xv * 0.7071067811865475f)); // exact GELU
        H[(size_t)m * DIM_D + n0 + ln] = (__bf16)g;
    }
}

// ---------------------------------------------------------------------------
// Stage 2: filt = H @ W2 + b2 (4096x1024x1024 GEMM), WMMA bf16, f32 out.
// Each wave computes 4 M-tiles (64 rows) per B fragment -> 4x less W2
// traffic, 4 back-to-back WMMAs per K step.
// ---------------------------------------------------------------------------
__global__ __launch_bounds__(128)
void filter_mlp2(const __bf16* __restrict__ H, const float* __restrict__ W2,
                 const float* __restrict__ b2, float* __restrict__ filt) {
    const int lane = threadIdx.x & 31;
    const int wave = threadIdx.x >> 5;
    const int h = lane >> 4, ln = lane & 15;
    const int n0 = blockIdx.x * 64 + wave * 16;
    const int m0 = blockIdx.y * 64;                 // 4 M-tiles of 16

    v8f acc[4] = {{}, {}, {}, {}};
    for (int kb = 0; kb < DIM_D; kb += 32) {
        v16bf b;
        #pragma unroll
        for (int e = 0; e < 16; ++e)
            b[e] = (__bf16)W2[(size_t)(kb + koffB(e, h)) * DIM_D + n0 + ln];
        #pragma unroll
        for (int t = 0; t < 4; ++t) {
            const __bf16* arow = H + (size_t)(m0 + t * 16 + ln) * DIM_D + kb;
            v16bf a;
            #pragma unroll
            for (int e = 0; e < 16; ++e) a[e] = arow[koffA(e, h)];
            acc[t] = __builtin_amdgcn_wmma_f32_16x16x32_bf16(false, a, false, b,
                                                             (short)0, acc[t],
                                                             false, false);
        }
    }
    const float bias = b2[n0 + ln];
    #pragma unroll
    for (int t = 0; t < 4; ++t)
        #pragma unroll
        for (int r = 0; r < 8; ++r)
            filt[(size_t)(m0 + t * 16 + r + 8 * h) * DIM_D + n0 + ln] = acc[t][r] + bias;
}

// ---------------------------------------------------------------------------
// In-LDS radix-2 DIT FFT, 8192 points, 256 threads. Input in bit-reversed
// order, output natural order. sign = -1 forward, +1 inverse (unnormalized).
// ---------------------------------------------------------------------------
__device__ __forceinline__ void fft8192(float2* buf, int tid, float sign) {
    #pragma unroll 1
    for (int stage = 0; stage < 13; ++stage) {
        const int m = 1 << stage;
        for (int j = tid; j < FFT_N / 2; j += 256) {
            int pos = j & (m - 1);
            int k0  = ((j >> stage) << (stage + 1)) + pos;
            int k1  = k0 + m;
            float ang = sign * 3.14159265358979323846f * (float)pos / (float)m;
            float sn, cs;
            __sincosf(ang, &sn, &cs);
            float2 a = buf[k0], bv = buf[k1];
            float2 t = make_float2(bv.x * cs - bv.y * sn, bv.x * sn + bv.y * cs);
            buf[k0] = make_float2(a.x + t.x, a.y + t.y);
            buf[k1] = make_float2(a.x - t.x, a.y - t.y);
        }
        __syncthreads();
    }
}

__device__ __forceinline__ float2 cmul(float2 a, float2 b) {
    return make_float2(a.x * b.x - a.y * b.y, a.x * b.y + a.y * b.x);
}

// Async global -> LDS staging of one f32 channel pair at bit-reversed LDS
// positions. buf is the kernel's only LDS allocation, so LDS byte addr of
// buf[i] is simply i*8. Uniform trip count (4096/256) -> no divergence.
__device__ __forceinline__ void async_load_bitrev(const float* row, int tid) {
    const uint64_t base = (uint64_t)(uintptr_t)row;
    for (int l = tid; l < SEQ_L; l += 256) {
        unsigned lds_addr = brev13((unsigned)l) * 8u;
        uint64_t ga = base + (uint64_t)l * (DIM_D * sizeof(float));
        asm volatile("global_load_async_to_lds_b64 %0, %1, off"
                     :: "v"(lds_addr), "v"(ga) : "memory");
    }
}

// ---------------------------------------------------------------------------
// Stage 3: filter spectra. One block per channel pair: z = f_d + i*f_{d+1},
// one complex FFT, Hermitian separation, store rfft bins k=0..4096 with the
// 1/N inverse-transform scale pre-folded in.
// ---------------------------------------------------------------------------
__global__ __launch_bounds__(256)
void filter_fft(const float* __restrict__ filt, float2* __restrict__ F) {
    extern __shared__ float2 buf[];
    const int d0  = 2 * blockIdx.x;
    const int tid = threadIdx.x;

    async_load_bitrev(filt + d0, tid);
    for (int l = SEQ_L + tid; l < FFT_N; l += 256)          // zero padding
        buf[brev13((unsigned)l)] = make_float2(0.f, 0.f);
    asm volatile("s_wait_asynccnt 0x0" ::: "memory");
    __syncthreads();

    fft8192(buf, tid, -1.0f);

    const float sc = 0.5f / (float)FFT_N;                   // separation * 1/N
    for (int k = tid; k <= FFT_N / 2; k += 256) {
        float2 z  = buf[k];
        float2 zm = buf[(FFT_N - k) & (FFT_N - 1)];
        float2 Fa = make_float2(sc * (z.x + zm.x),  sc * (z.y - zm.y));
        float2 Fb = make_float2(sc * (z.y + zm.y), -sc * (z.x - zm.x));
        F[(size_t)d0 * 4097 + k]       = Fa;
        F[(size_t)(d0 + 1) * 4097 + k] = Fb;
    }
}

// ---------------------------------------------------------------------------
// Stage 4: the convolution. One block per (batch, channel pair).
// async-load -> FFT -> spectral multiply (Hermitian combine) -> inverse FFT
// -> async LDS->global store (no scaling needed; folded into F).
// ---------------------------------------------------------------------------
__global__ __launch_bounds__(256)
void fft_conv(const float* __restrict__ x, const float2* __restrict__ F,
              float* __restrict__ out) {
    extern __shared__ float2 buf[];
    const int p   = blockIdx.x & 511;   // channel pair
    const int b   = blockIdx.x >> 9;    // batch
    const int d0  = 2 * p;
    const int tid = threadIdx.x;

    async_load_bitrev(x + (size_t)b * SEQ_L * DIM_D + d0, tid);
    for (int l = SEQ_L + tid; l < FFT_N; l += 256)
        buf[brev13((unsigned)l)] = make_float2(0.f, 0.f);
    asm volatile("s_wait_asynccnt 0x0" ::: "memory");
    __syncthreads();

    fft8192(buf, tid, -1.0f);

    // Spectral combine. Each k in [0,4096] and its mirror N-k in [4097,8191]
    // are touched only by the thread owning k -> no intra-step race.
    const float2* Fa_ptr = F + (size_t)d0 * 4097;
    const float2* Fb_ptr = F + (size_t)(d0 + 1) * 4097;
    for (int k = tid; k <= FFT_N / 2; k += 256) {
        int km = (FFT_N - k) & (FFT_N - 1);
        float2 z  = buf[k];
        float2 zm = buf[km];
        // Xa/Xb carry a factor of 2; the matching 0.5*(1/N) lives in F already.
        float2 Xa = make_float2(z.x + zm.x,  z.y - zm.y);
        float2 Xb = make_float2(z.y + zm.y, -(z.x - zm.x));
        float2 A  = cmul(Xa, Fa_ptr[k]);
        float2 Bc = cmul(Xb, Fb_ptr[k]);
        buf[k]  = make_float2(A.x - Bc.y,  A.y + Bc.x);   // A + i*B
        buf[km] = make_float2(A.x + Bc.y, -A.y + Bc.x);   // conj(A) + i*conj(B)
    }
    __syncthreads();

    // In-place bit-reverse permute (pair owned by smaller index).
    for (int k = tid; k < FFT_N; k += 256) {
        int r = (int)brev13((unsigned)k);
        if (r > k) { float2 t = buf[k]; buf[k] = buf[r]; buf[r] = t; }
    }
    __syncthreads();
    fft8192(buf, tid, 1.0f);                     // inverse (scale pre-folded)

    // Async LDS -> global store of the first L samples (both channels).
    const uint64_t obase = (uint64_t)(uintptr_t)(out + (size_t)b * SEQ_L * DIM_D + d0);
    for (int l = tid; l < SEQ_L; l += 256) {
        unsigned lds_addr = (unsigned)l * 8u;
        uint64_t ga = obase + (uint64_t)l * (DIM_D * sizeof(float));
        asm volatile("global_store_async_from_lds_b64 %0, %1, off"
                     :: "v"(ga), "v"(lds_addr) : "memory");
    }
    asm volatile("s_wait_asynccnt 0x0" ::: "memory");
}

// ---------------------------------------------------------------------------
extern "C" void kernel_launch(void* const* d_in, const int* in_sizes, int n_in,
                              void* d_out, int out_size, void* d_ws, size_t ws_size,
                              hipStream_t stream) {
    const float* x  = (const float*)d_in[0];
    const float* W1 = (const float*)d_in[1];
    const float* b1 = (const float*)d_in[2];
    const float* W2 = (const float*)d_in[3];
    const float* b2 = (const float*)d_in[4];
    float* out = (float*)d_out;

    char* ws = (char*)d_ws;
    float*  filt = (float*)ws;                            // 16 MB   @ 0
    __bf16* H    = (__bf16*)(ws + ((size_t)16 << 20));    //  8 MB   @ 16 MB (dead after mlp2)
    float2* F    = (float2*)(ws + ((size_t)16 << 20));    // 33.5 MB @ 16 MB (reuses H region)

    filter_mlp1<<<dim3(DIM_D / 64, SEQ_L / 16), 128, 0, stream>>>(W1, b1, H);
    filter_mlp2<<<dim3(DIM_D / 64, SEQ_L / 64), 128, 0, stream>>>(H, W2, b2, filt);

    const size_t lds_bytes = FFT_N * sizeof(float2);      // 64 KB
    filter_fft<<<DIM_D / 2, 256, lds_bytes, stream>>>(filt, F);
    fft_conv<<<4 * (DIM_D / 2), 256, lds_bytes, stream>>>(x, F, out);
}
